// Attention_19129784336707
// MI455X (gfx1250) — matrix operand order repaired and checked by reading, not compiled, not run on hardware
//
#include <hip/hip_runtime.h>

typedef __attribute__((ext_vector_type(16))) __bf16 v16bf;
typedef __attribute__((ext_vector_type(8)))  __bf16 bf16x8;
typedef __attribute__((ext_vector_type(4)))  __bf16 bf16x4;
typedef __attribute__((ext_vector_type(8)))  float  v8f;

#define BATCH   16
#define S_LEN   4096
#define D       512
#define MBLK    128      // s-rows per block (8 waves x 16)
#define NCHUNK  64       // h-columns per chunk (4 accumulator frags / wave)
#define APITCH  520      // padded LDS row pitch in bf16 elems (16B-aligned, bank-spread)

// gfx1250 hardware tanh (V_TANH_F32) if the toolchain exposes it
#if __has_builtin(__builtin_amdgcn_tanhf)
static __device__ __forceinline__ float fast_tanh(float x) { return __builtin_amdgcn_tanhf(x); }
#else
static __device__ __forceinline__ float fast_tanh(float x) { return tanhf(x); }
#endif

// ---------------- kernel: f32 -> bf16 convert (W_ctx) ----------------
__global__ __launch_bounds__(256) void k_cvt_bf16(const float* __restrict__ w,
                                                  __bf16* __restrict__ o, int n) {
    int i = blockIdx.x * 256 + threadIdx.x;
    if (i < n) o[i] = (__bf16)w[i];
}

// ---------------- kernel: y[b,h] = bias[h] + sum_i x[b,i]*W[h,i] ------
__global__ __launch_bounds__(256) void k_linear16(const float* __restrict__ x,
                                                  const float* __restrict__ W,
                                                  const float* __restrict__ bias,
                                                  float* __restrict__ y) {
    __shared__ float sx[D];
    int b = blockIdx.y;
    int h = blockIdx.x * 256 + threadIdx.x;
    for (int i = threadIdx.x; i < D; i += 256) sx[i] = x[b * D + i];
    __syncthreads();
    float acc = bias[h];
    const float* wr = W + (size_t)h * D;
    #pragma unroll 8
    for (int i = 0; i < D; ++i) acc = fmaf(sx[i], wr[i], acc);
    y[b * D + h] = acc;
}

// ---------------- main WMMA kernel: fused ctx-GEMM + additive scores ----
// scores[b,s] = mask ? -inf : sum_h v[h]*tanh(inp[b,h]+b_ctx[h] + (ctx@Wc^T)[s,h])
__global__ __launch_bounds__(256) void k_scores_wmma(
        const float* __restrict__ ctx,          // [B,S,D] f32
        const unsigned char* __restrict__ mask, // [B,S] bool
        const __bf16* __restrict__ wc,          // [D,D] bf16 (W_ctx row-major)
        const float* __restrict__ inp,          // [B,D] f32
        const float* __restrict__ bctx,         // [D]
        const float* __restrict__ vvec,         // [D]
        float* __restrict__ scores)             // [B,S]
{
    extern __shared__ char smem[];
    __bf16* sA  = (__bf16*)smem;                                      // [MBLK][APITCH]
    __bf16* sB  = (__bf16*)(smem + MBLK * APITCH * 2);                // [NCHUNK][APITCH]
    float*  sC1 = (float*)(smem + (MBLK + NCHUNK) * APITCH * 2);      // [D] inp+b_ctx
    float*  sV  = sC1 + D;                                            // [D] v

    const int tid  = threadIdx.x;
    const int lane = tid & 31;
    const int wid  = tid >> 5;
    const int ml   = lane & 15;
    const int half = lane >> 4;
    const int b    = blockIdx.y;
    const int s0   = blockIdx.x * MBLK;

    // stage combined bias and v
    for (int i = tid; i < D; i += 256) {
        sC1[i] = inp[b * D + i] + bctx[i];
        sV[i]  = vvec[i];
    }
    // stage A tile: context[b, s0:s0+128, :] f32 -> bf16 LDS.
    // Batched: 8 global_load_b128 in flight before the first LDS store.
    const float* cbase = ctx + ((size_t)(b * S_LEN + s0)) * D;
    #pragma unroll
    for (int ot = 0; ot < 8; ++ot) {
        float4 f[8];
        #pragma unroll
        for (int ii = 0; ii < 8; ++ii) {
            int idx = ((ot * 8 + ii) * 256 + tid) * 4;
            int r = idx >> 9, k = idx & (D - 1);
            f[ii] = *(const float4*)(cbase + (size_t)r * D + k);
        }
        #pragma unroll
        for (int ii = 0; ii < 8; ++ii) {
            int idx = ((ot * 8 + ii) * 256 + tid) * 4;
            int r = idx >> 9, k = idx & (D - 1);
            bf16x4 o;
            o[0] = (__bf16)f[ii].x; o[1] = (__bf16)f[ii].y;
            o[2] = (__bf16)f[ii].z; o[3] = (__bf16)f[ii].w;
            *(bf16x4*)(sA + r * APITCH + k) = o;
        }
    }
    __syncthreads();

    float sc[8];
    #pragma unroll
    for (int j = 0; j < 8; ++j) sc[j] = 0.f;

    const int aRow = wid * 16 + ml;

    for (int nc = 0; nc < D / NCHUNK; ++nc) {               // 8 column chunks
        // stage B chunk: wc rows [nc*64, nc*64+64) -> LDS (batched loads)
        #pragma unroll
        for (int ot = 0; ot < 2; ++ot) {
            bf16x8 w[8];
            #pragma unroll
            for (int ii = 0; ii < 8; ++ii) {
                int idx = ((ot * 8 + ii) * 256 + tid) * 8;
                int n = idx >> 9, k = idx & (D - 1);
                w[ii] = *(const bf16x8*)(wc + (size_t)(nc * NCHUNK + n) * D + k);
            }
            #pragma unroll
            for (int ii = 0; ii < 8; ++ii) {
                int idx = ((ot * 8 + ii) * 256 + tid) * 8;
                int n = idx >> 9, k = idx & (D - 1);
                *(bf16x8*)(sB + n * APITCH + k) = w[ii];
            }
        }
        __syncthreads();

        v8f acc[4];
        #pragma unroll
        for (int q = 0; q < 4; ++q) acc[q] = (v8f){};

        #pragma unroll 2
        for (int ks = 0; ks < D / 32; ++ks) {               // 16 K-steps
            int kb = ks * 32;
            // A frag (16x32 bf16): elems 0..7 -> k = kb+8h+0..7 ; 8..15 -> kb+16+8h+0..7
            const __bf16* ap = sA + aRow * APITCH + kb + half * 8;
            bf16x8 a0 = *(const bf16x8*)ap;
            bf16x8 a1 = *(const bf16x8*)(ap + 16);
            v16bf af = __builtin_shufflevector(a0, a1, 0,1,2,3,4,5,6,7,8,9,10,11,12,13,14,15);
            // Preload ALL four B frags into distinct registers so the scheduler
            // can keep 8 ds_load_b128 in flight and gate each WMMA with a
            // partial s_wait_dscnt instead of a full drain.
            v16bf bf[4];
            #pragma unroll
            for (int q = 0; q < 4; ++q) {
                const __bf16* bp = sB + (q * 16 + ml) * APITCH + kb + half * 8;
                bf16x8 bl = *(const bf16x8*)bp;
                bf16x8 bh = *(const bf16x8*)(bp + 16);
                bf[q] = __builtin_shufflevector(bl, bh, 0,1,2,3,4,5,6,7,8,9,10,11,12,13,14,15);
            }
            #pragma unroll
            for (int q = 0; q < 4; ++q)
                acc[q] = __builtin_amdgcn_wmma_f32_16x16x32_bf16(false, af, false, bf[q],
                                                                 (short)0, acc[q], false, false);
        }
        // fold this 64-column chunk into the additive scores (HW v_tanh_f32)
        #pragma unroll
        for (int q = 0; q < 4; ++q) {
            int n = nc * NCHUNK + q * 16 + ml;
            float c1 = sC1[n], vv = sV[n];
            #pragma unroll
            for (int j = 0; j < 8; ++j)
                sc[j] += vv * fast_tanh(c1 + acc[q][j]);    // row m = j + 8*half
        }
        __syncthreads();   // before next chunk overwrites sB
    }

    // reduce over the 16 lanes holding different n residues
    #pragma unroll
    for (int m = 1; m < 16; m <<= 1)
        #pragma unroll
        for (int j = 0; j < 8; ++j) sc[j] += __shfl_xor(sc[j], m, 32);

    if (ml == 0) {
        int rbase = s0 + wid * 16 + half * 8;    // half0 -> rows 0..7, half1 -> 8..15
        #pragma unroll
        for (int j = 0; j < 8; ++j) {
            int s = rbase + j;
            scores[b * S_LEN + s] = mask[(size_t)b * S_LEN + s] ? -__builtin_inff() : sc[j];
        }
    }
}

// ---------------- masked softmax per batch (in place) -----------------
__global__ __launch_bounds__(256) void k_softmax(float* __restrict__ attn) {
    __shared__ float red[256];
    int b = blockIdx.x, tid = threadIdx.x;
    float* row = attn + (size_t)b * S_LEN;
    float mx = -__builtin_inff();
    for (int s = tid; s < S_LEN; s += 256) mx = fmaxf(mx, row[s]);
    red[tid] = mx; __syncthreads();
    for (int o = 128; o > 0; o >>= 1) { if (tid < o) red[tid] = fmaxf(red[tid], red[tid + o]); __syncthreads(); }
    mx = red[0]; __syncthreads();
    float sum = 0.f;
    for (int s = tid; s < S_LEN; s += 256) { float e = __expf(row[s] - mx); row[s] = e; sum += e; }
    red[tid] = sum; __syncthreads();
    for (int o = 128; o > 0; o >>= 1) { if (tid < o) red[tid] += red[tid + o]; __syncthreads(); }
    float inv = 1.f / red[0];
    for (int s = tid; s < S_LEN; s += 256) row[s] *= inv;
}

// ---------------- u[b,i] = sum_s attn[b,s] * context[b,s,i] -----------
__global__ __launch_bounds__(128) void k_attn_ctx(const float* __restrict__ ctx,
                                                  const float* __restrict__ attn,
                                                  float* __restrict__ u) {
    int b = blockIdx.y;
    int i = blockIdx.x * 128 + threadIdx.x;
    const float* cb = ctx + (size_t)b * S_LEN * D + i;
    const float* ab = attn + (size_t)b * S_LEN;
    float a0 = 0.f, a1 = 0.f, a2 = 0.f, a3 = 0.f;
    for (int s = 0; s < S_LEN; s += 4) {
        a0 = fmaf(ab[s + 0], cb[(size_t)(s + 0) * D], a0);
        a1 = fmaf(ab[s + 1], cb[(size_t)(s + 1) * D], a1);
        a2 = fmaf(ab[s + 2], cb[(size_t)(s + 2) * D], a2);
        a3 = fmaf(ab[s + 3], cb[(size_t)(s + 3) * D], a3);
    }
    u[b * D + i] = (a0 + a1) + (a2 + a3);
}

extern "C" void kernel_launch(void* const* d_in, const int* in_sizes, int n_in,
                              void* d_out, int out_size, void* d_ws, size_t ws_size,
                              hipStream_t stream) {
    const float*         input  = (const float*)d_in[0];         // [16,512]
    const float*         ctx    = (const float*)d_in[1];         // [16,4096,512]
    const unsigned char* mask   = (const unsigned char*)d_in[2]; // [16,4096] bool
    const float*         W_in   = (const float*)d_in[3];         // [512,512]
    const float*         b_in   = (const float*)d_in[4];         // [512]
    const float*         W_ctx  = (const float*)d_in[5];         // [512,512]
    const float*         b_ctx  = (const float*)d_in[6];         // [512]
    const float*         v      = (const float*)d_in[7];         // [512]

    float* hidden = (float*)d_out;                       // [16,512]
    float* attn   = (float*)d_out + BATCH * D;           // [16,4096]

    __bf16* wcbf = (__bf16*)d_ws;                                // 512 KB
    float*  inp  = (float*)((char*)d_ws + (size_t)D * D * 2);    // [16,512]
    float*  u    = inp + BATCH * D;                              // [16,512]

    // 1) W_ctx -> bf16 (L2-resident operand for WMMA)
    k_cvt_bf16<<<dim3((D * D) / 256), dim3(256), 0, stream>>>(W_ctx, wcbf, D * D);
    // 2) inp = input @ W_in^T + b_in
    k_linear16<<<dim3(D / 256, BATCH), dim3(256), 0, stream>>>(input, W_in, b_in, inp);
    // 3) fused WMMA ctx-projection + additive attention scores
    size_t smem = (size_t)(MBLK + NCHUNK) * APITCH * 2 + 2 * D * 4;   // ~204 KB
    k_scores_wmma<<<dim3(S_LEN / MBLK, BATCH), dim3(256), smem, stream>>>(
        ctx, mask, wcbf, inp, b_ctx, v, attn);
    // 4) masked softmax (in place in attn region of d_out)
    k_softmax<<<dim3(BATCH), dim3(256), 0, stream>>>(attn);
    // 5) u = attn^T . context   (streaming, memory-bound)
    k_attn_ctx<<<dim3(D / 128, BATCH), dim3(128), 0, stream>>>(ctx, attn, u);
    // 6) hidden = u @ W_ctx^T + b_ctx  (valid since sum(attn)=1)
    k_linear16<<<dim3(D / 256, BATCH), dim3(256), 0, stream>>>(u, W_ctx, b_ctx, hidden);
}
